// GTLayer_15496242004781
// MI455X (gfx1250) — matrix operand order repaired and checked by reading, not compiled, not run on hardware
//
#include <hip/hip_runtime.h>

// Problem constants (from the reference)
#define NDIM  4096
#define C_IN  5
#define C_OUT 2
static const size_t NN = (size_t)NDIM * (size_t)NDIM;

typedef __attribute__((ext_vector_type(16))) _Float16 v16h;
typedef __attribute__((ext_vector_type(8)))  _Float16 v8h;
typedef __attribute__((ext_vector_type(4)))  _Float16 v4h;
typedef __attribute__((ext_vector_type(8)))  float    v8f;
typedef __attribute__((ext_vector_type(4)))  float    v4f;

// ---------------------------------------------------------------------------
// Kernel 1: softmax over C_IN for both weight matrices.
// Writes f1/f2 into workspace AND into the tail of d_out.
// ---------------------------------------------------------------------------
__global__ void gt_softmax(const float* __restrict__ w1,
                           const float* __restrict__ w2,
                           float* __restrict__ fws,
                           float* __restrict__ out_tail) {
  int t = threadIdx.x;
  if (t < 2 * C_OUT) {
    const float* w = (t < C_OUT) ? w1 : w2;
    int row = t & 1;
    const float* wr = w + row * C_IN;
    float m = wr[0];
    #pragma unroll
    for (int j = 1; j < C_IN; ++j) m = fmaxf(m, wr[j]);
    float e[C_IN];
    float s = 0.f;
    #pragma unroll
    for (int j = 0; j < C_IN; ++j) { e[j] = __expf(wr[j] - m); s += e[j]; }
    float inv = 1.f / s;
    #pragma unroll
    for (int j = 0; j < C_IN; ++j) {
      float v = e[j] * inv;
      fws[t * C_IN + j]      = v;
      out_tail[t * C_IN + j] = v;
    }
  }
}

// ---------------------------------------------------------------------------
// Kernel 2: tiled (64x64) weighted sum. One pass over A produces
//   sA [c][n][m]  row-major fp16                   (GEMM A operand)
//   sBT[c][m][n]  = sumB[c][n][m] transposed fp16  (GEMM B operand, N x K)
// Transpose goes through an LDS staging tile; all global traffic coalesced.
// ---------------------------------------------------------------------------
#define T2 64
__global__ __launch_bounds__(256) void gt_wsum(const float* __restrict__ A,
                                               const float* __restrict__ fws,
                                               _Float16* __restrict__ sA,
                                               _Float16* __restrict__ sBT) {
  __shared__ _Float16 tB[C_OUT][T2][T2 + 8];

  const int tx = blockIdx.x;   // column tile (fast dim m)
  const int ty = blockIdx.y;   // row tile (n)
  const int t  = threadIdx.x;

  float f1[C_OUT][C_IN], f2[C_OUT][C_IN];
  #pragma unroll
  for (int c = 0; c < C_OUT; ++c)
    #pragma unroll
    for (int j = 0; j < C_IN; ++j) {
      f1[c][j] = fws[c * C_IN + j];
      f2[c][j] = fws[2 * C_IN + c * C_IN + j];
    }

  // 64x64 f32 tile = 1024 float4 chunks; 256 threads x 4 iters.
  #pragma unroll
  for (int i = 0; i < 4; ++i) {
    int idx = t + i * 256;
    int r = idx >> 4;            // 0..63 row within tile
    int c4 = idx & 15;           // 0..15 float4 column
    size_t goff = (size_t)(ty * T2 + r) * NDIM + (size_t)(tx * T2 + c4 * 4);

    v4f acc1[C_OUT] = {}, acc2[C_OUT] = {};
    #pragma unroll
    for (int j = 0; j < C_IN; ++j) {
      v4f a = *(const v4f*)(A + (size_t)j * NN + goff);
      #pragma unroll
      for (int c = 0; c < C_OUT; ++c) {
        acc1[c] += f1[c][j] * a;
        acc2[c] += f2[c][j] * a;
      }
    }
    #pragma unroll
    for (int c = 0; c < C_OUT; ++c) {
      v4h h1;
      #pragma unroll
      for (int k = 0; k < 4; ++k) h1[k] = (_Float16)acc1[c][k];
      *(v4h*)(sA + (size_t)c * NN + goff) = h1;           // coalesced
      #pragma unroll
      for (int k = 0; k < 4; ++k)
        tB[c][r][c4 * 4 + k] = (_Float16)acc2[c][k];      // stage for transpose
    }
  }
  __syncthreads();

  // write sBT: row = m (tx tile), col = n (ty tile); coalesced v4h stores
  #pragma unroll
  for (int i = 0; i < 4; ++i) {
    int idx = t + i * 256;
    int rm = idx >> 4;           // 0..63 row of sBT within tile (= original col)
    int cn4 = idx & 15;          // 0..15 v4 column (= original row / 4)
    size_t goff = (size_t)(tx * T2 + rm) * NDIM + (size_t)(ty * T2 + cn4 * 4);
    #pragma unroll
    for (int c = 0; c < C_OUT; ++c) {
      v4h h;
      #pragma unroll
      for (int k = 0; k < 4; ++k) h[k] = tB[c][cn4 * 4 + k][rm];
      *(v4h*)(sBT + (size_t)c * NN + goff) = h;
    }
  }
}

// ---------------------------------------------------------------------------
// Kernel 3: per-channel GEMM H[c] = sA[c] (MxK row-major f16)
//                                  x sBT[c]^T (sBT is NxK row-major f16).
// 128x128x64 block tiles, 8 wave32s (4Mx2N), 2x4 WMMA accums per wave,
// 2 K-substeps (16 WMMAs) per barrier. Double-buffered LDS fed by gfx1250
// async-to-LDS copies (ASYNCcnt).
// ---------------------------------------------------------------------------
#define BM 128
#define BN 128
#define BK 64
#define LDS_STRIDE (BK + 8)   // halves; padded against bank conflicts

__global__ __launch_bounds__(256) void gt_gemm(const _Float16* __restrict__ Am,
                                               const _Float16* __restrict__ Bt,
                                               float* __restrict__ H) {
  __shared__ _Float16 As[2][BM * LDS_STRIDE];   // [row m][k]
  __shared__ _Float16 Bs[2][BN * LDS_STRIDE];   // [row n][k]

  const int c = blockIdx.z;
  const size_t chOff = (size_t)c * NN;
  const _Float16* Ab = Am + chOff + (size_t)blockIdx.y * BM * NDIM;
  const _Float16* Bb = Bt + chOff + (size_t)blockIdx.x * BN * NDIM;
  float* Hb = H + chOff + (size_t)blockIdx.y * BM * NDIM + (size_t)blockIdx.x * BN;

  const int t    = threadIdx.x;
  const int lane = t & 31;
  const int wave = t >> 5;     // 0..7
  const int wm   = wave >> 1;  // 0..3  (M direction)
  const int wn   = wave & 1;   // 0..1  (N direction)
  const int l16  = lane & 15;
  const int lhi  = lane >> 4;  // 0 or 1

  v8f acc[2][4] = {};

  // Issue async global->LDS copies for one 128x64-half K-tile of A and BT:
  // 1024 16-byte chunks each; 256 threads x 4 iters x 2 matrices.
  auto stage = [&](int k0, int buf) {
    #pragma unroll
    for (int it = 0; it < 4; ++it) {
      int idx = t + it * 256;            // 0..1023
      int r   = idx >> 3;                // row 0..127 (8 chunks per row)
      int ck  = (idx & 7) * 8;           // half offset 0..56
      unsigned long long ga =
          (unsigned long long)(uintptr_t)(Ab + (size_t)r * NDIM + (k0 + ck));
      unsigned int la =
          (unsigned int)(uintptr_t)&As[buf][r * LDS_STRIDE + ck];
      asm volatile("global_load_async_to_lds_b128 %0, %1, off"
                   :: "v"(la), "v"(ga) : "memory");
      unsigned long long gb =
          (unsigned long long)(uintptr_t)(Bb + (size_t)r * NDIM + (k0 + ck));
      unsigned int lb =
          (unsigned int)(uintptr_t)&Bs[buf][r * LDS_STRIDE + ck];
      asm volatile("global_load_async_to_lds_b128 %0, %1, off"
                   :: "v"(lb), "v"(gb) : "memory");
    }
  };

  stage(0, 0);
  asm volatile("s_wait_asynccnt 0x0" ::: "memory");
  __syncthreads();

  int cur = 0;
  for (int k0 = 0; k0 < NDIM; k0 += BK, cur ^= 1) {
    if (k0 + BK < NDIM) stage(k0 + BK, cur ^ 1);

    const _Float16* Ac = &As[cur][0];
    const _Float16* Bc = &Bs[cur][0];

    // Two K=32 substeps per staged tile: 16 WMMAs per barrier.
    #pragma unroll
    for (int ks = 0; ks < BK; ks += 32) {
      v16h afrag[2], bfrag[4];
      #pragma unroll
      for (int mt = 0; mt < 2; ++mt) {
        // A 16x32 layout: lanes 0-15 K{0..7,16..23}; lanes 16-31 K{8..15,24..31}
        int row = wm * 32 + mt * 16 + l16;
        int k1  = ks + lhi * 8;
        v8h lo = *(const v8h*)(Ac + row * LDS_STRIDE + k1);
        v8h hi = *(const v8h*)(Ac + row * LDS_STRIDE + k1 + 16);
        afrag[mt] = __builtin_shufflevector(lo, hi,
                    0,1,2,3,4,5,6,7,8,9,10,11,12,13,14,15);
      }
      #pragma unroll
      for (int nt = 0; nt < 4; ++nt) {
        // B 32x16 layout: lane = column; lanes 0-15 K{0..7,16..23}; lanes 16-31 K{8..15,24..31}
        int col = wn * 64 + nt * 16 + l16;
        int k2  = ks + lhi * 8;
        v8h lo = *(const v8h*)(Bc + col * LDS_STRIDE + k2);
        v8h hi = *(const v8h*)(Bc + col * LDS_STRIDE + k2 + 16);
        bfrag[nt] = __builtin_shufflevector(lo, hi,
                    0,1,2,3,4,5,6,7,8,9,10,11,12,13,14,15);
      }

      #pragma unroll
      for (int mt = 0; mt < 2; ++mt)
        #pragma unroll
        for (int nt = 0; nt < 4; ++nt)
          acc[mt][nt] = __builtin_amdgcn_wmma_f32_16x16x32_f16(
              false, afrag[mt], false, bfrag[nt],
              (short)0, acc[mt][nt], false, false);
    }

    // Own async copies done; barrier also flushes LDS reads (s_wait_dscnt 0)
    // so the buffer we just read can be overwritten next iteration.
    asm volatile("s_wait_asynccnt 0x0" ::: "memory");
    __syncthreads();
  }

  // Epilogue: C/D layout -> lanes 0-15 M=r, lanes 16-31 M=r+8
  #pragma unroll
  for (int mt = 0; mt < 2; ++mt) {
    #pragma unroll
    for (int nt = 0; nt < 4; ++nt) {
      int col = wn * 64 + nt * 16 + l16;
      #pragma unroll
      for (int r = 0; r < 8; ++r) {
        int row = wm * 32 + mt * 16 + lhi * 8 + r;
        Hb[(size_t)row * NDIM + col] = acc[mt][nt][r];
      }
    }
  }
}

// ---------------------------------------------------------------------------
// Host-side launcher
// inputs:  d_in[0]=A [5,4096,4096] f32, d_in[1]=weight1 [2,5], d_in[2]=weight2 [2,5]
// output:  d_out = [H (2*N*N) | f1 (10) | f2 (10)] f32
// scratch: d_ws  = [fws 20 f32 (pad 256B) | sA 2*N*N f16 | sBT 2*N*N f16]
// ---------------------------------------------------------------------------
extern "C" void kernel_launch(void* const* d_in, const int* in_sizes, int n_in,
                              void* d_out, int out_size, void* d_ws, size_t ws_size,
                              hipStream_t stream) {
  const float* A  = (const float*)d_in[0];
  const float* w1 = (const float*)d_in[1];
  const float* w2 = (const float*)d_in[2];

  float*    out_f = (float*)d_out;
  float*    tail  = out_f + (size_t)C_OUT * NN;

  float*    fws = (float*)d_ws;
  _Float16* sA  = (_Float16*)((char*)d_ws + 256);
  _Float16* sBT = sA + (size_t)C_OUT * NN;

  gt_softmax<<<1, 32, 0, stream>>>(w1, w2, fws, tail);

  {
    dim3 grid(NDIM / T2, NDIM / T2);
    gt_wsum<<<grid, 256, 0, stream>>>(A, fws, sA, sBT);
  }

  {
    dim3 grid(NDIM / BN, NDIM / BM, C_OUT);
    gt_gemm<<<grid, 256, 0, stream>>>(sA, sBT, out_f);
  }
}
